// SimpleGPT_54503134986924
// MI455X (gfx1250) — matrix operand order repaired and checked by reading, not compiled, not run on hardware
//
#include <hip/hip_runtime.h>
#include <hip/hip_bf16.h>
#include <stdint.h>

// Problem constants (match reference)
#define B_  8
#define T_  512
#define E_  256
#define H_  512
#define V_  32000
#define M4  (B_ * T_)   // 4096 rows (b*T + t)
#define G4  (4 * H_)    // 2048 gate columns

typedef unsigned short u16;
typedef __attribute__((ext_vector_type(16))) __bf16 v16bf;
typedef __attribute__((ext_vector_type(8)))  float  v8f;

union FragBF { uint4 u[2]; v16bf v; };

__device__ __forceinline__ u16 f32_to_bf16(float f) {
  unsigned u = __float_as_uint(f);
  unsigned r = 0x7FFFu + ((u >> 16) & 1u);
  return (u16)((u + r) >> 16);
}

// ---- WMMA fragment loaders (contiguous k = 16*(lane/16) + s in both A and B, wave32) ----
// A (16x32, MxK): lanes 0-15 -> M=lane, K=[0..15]; lanes 16-31 -> K=[16..31]
__device__ __forceinline__ v16bf load_a_frag(const u16* __restrict__ A, int K,
                                             int mbase, int kc) {
  int lane = threadIdx.x & 31;
  int m = mbase + (lane & 15);
  int b0 = (lane >= 16) ? 16 : 0;
  const u16* p = A + (size_t)m * K + kc * 32 + b0;
  FragBF f;
  f.u[0] = *(const uint4*)(p);       // K = kc*32 + b0 + [0..7]
  f.u[1] = *(const uint4*)(p + 8);   // K = kc*32 + b0 + [8..15]
  return f.v;
}

// B (32x16, KxN) from weights stored row-major [N,K]:
// lanes 0-15 -> N=lane, K=[0..15]; lanes 16-31 -> N=lane-16, K=[16..31]
__device__ __forceinline__ v16bf load_b_frag(const u16* __restrict__ W, int K,
                                             int nbase, int kc) {
  int lane = threadIdx.x & 31;
  int n = nbase + (lane & 15);
  const u16* p = W + (size_t)n * K + kc * 32 + ((lane >= 16) ? 16 : 0);
  FragBF f;
  f.u[0] = *(const uint4*)(p);
  f.u[1] = *(const uint4*)(p + 8);
  return f.v;
}

// ---- fp32 -> bf16 conversion (vectorized x4) ----
__global__ __launch_bounds__(256) void cvt_bf16_kernel(const float* __restrict__ in,
                                                       ushort4* __restrict__ out,
                                                       int n4) {
  int i = blockIdx.x * 256 + threadIdx.x;
  if (i < n4) {
    float4 f = ((const float4*)in)[i];
    ushort4 o;
    o.x = f32_to_bf16(f.x);
    o.y = f32_to_bf16(f.y);
    o.z = f32_to_bf16(f.z);
    o.w = f32_to_bf16(f.w);
    out[i] = o;
  }
}

// ---- embedding gather -> bf16 [M4, E_] ----
__global__ __launch_bounds__(256) void embed_kernel(const int* __restrict__ ids,
                                                    const float* __restrict__ emb,
                                                    u16* __restrict__ x0) {
  int i = blockIdx.x * 256 + threadIdx.x;   // i in [0, M4*E_)
  int row = i >> 8;          // E_ = 256
  int e   = i & 255;
  int id  = ids[row];
  x0[i] = f32_to_bf16(emb[(size_t)id * E_ + e]);
}

// ---- generic bf16 GEMM: C[M,N] = A[M,K] * W[N,K]^T + bias0 (+ bias1) ----
// block = 256 threads (8 waves); tile 128x128; wave tile 32x64 (2x4 wmma tiles)
// NT = non-temporal output stores (use for the write-once logits tensor)
template <bool NT>
__global__ __launch_bounds__(256) void gemm_bf16_kernel(
    const u16* __restrict__ A, const u16* __restrict__ W,
    const float* __restrict__ bias0, const float* __restrict__ bias1,
    float* __restrict__ C, int M, int N, int K) {
  const int wave = threadIdx.x >> 5;
  const int lane = threadIdx.x & 31;
  const int wm = wave >> 1;                 // 0..3
  const int wn = wave & 1;                  // 0..1
  const int mblk = blockIdx.y * 128 + wm * 32;
  const int nblk = blockIdx.x * 128 + wn * 64;

  v8f zero = {0.f, 0.f, 0.f, 0.f, 0.f, 0.f, 0.f, 0.f};
  v8f acc[2][4];
#pragma unroll
  for (int i = 0; i < 2; i++)
#pragma unroll
    for (int j = 0; j < 4; j++) acc[i][j] = zero;

  const int nk = K >> 5;
  for (int kc = 0; kc < nk; ++kc) {
    v16bf a0 = load_a_frag(A, K, mblk, kc);
    v16bf a1 = load_a_frag(A, K, mblk + 16, kc);
#pragma unroll
    for (int j = 0; j < 4; j++) {
      v16bf b = load_b_frag(W, K, nblk + j * 16, kc);
      acc[0][j] = __builtin_amdgcn_wmma_f32_16x16x32_bf16(
          false, a0, false, b, (short)0, acc[0][j], false, false);
      acc[1][j] = __builtin_amdgcn_wmma_f32_16x16x32_bf16(
          false, a1, false, b, (short)0, acc[1][j], false, false);
    }
  }

  // epilogue: C/D layout -> VGPR v: row = v + (lane>=16?8:0); col = lane&15
  const int rofs = (lane >= 16) ? 8 : 0;
#pragma unroll
  for (int i = 0; i < 2; i++) {
#pragma unroll
    for (int j = 0; j < 4; j++) {
      int n = nblk + j * 16 + (lane & 15);
      float bv = bias0 ? bias0[n] : 0.f;
      if (bias1) bv += bias1[n];
      int mb = mblk + i * 16 + rofs;
#pragma unroll
      for (int v = 0; v < 8; v++) {
        float val = acc[i][j][v] + bv;
        float* dst = &C[(size_t)(mb + v) * N + n];
        if (NT) {
          __builtin_nontemporal_store(val, dst);
        } else {
          *dst = val;
        }
      }
    }
  }
}

// ---- software grid barrier (agent-scope atomics) ----
__device__ __forceinline__ void grid_barrier(unsigned* bar) {
  __syncthreads();
  __threadfence();
  if (threadIdx.x == 0) {
    unsigned* cnt = bar;
    unsigned* phase = bar + 1;
    unsigned my = __hip_atomic_load(phase, __ATOMIC_ACQUIRE, __HIP_MEMORY_SCOPE_AGENT);
    unsigned prev = __hip_atomic_fetch_add(cnt, 1u, __ATOMIC_ACQ_REL, __HIP_MEMORY_SCOPE_AGENT);
    if (prev == gridDim.x - 1) {
      __hip_atomic_store(cnt, 0u, __ATOMIC_RELAXED, __HIP_MEMORY_SCOPE_AGENT);
      __hip_atomic_fetch_add(phase, 1u, __ATOMIC_ACQ_REL, __HIP_MEMORY_SCOPE_AGENT);
    } else {
      while (__hip_atomic_load(phase, __ATOMIC_ACQUIRE, __HIP_MEMORY_SCOPE_AGENT) == my) {
        __builtin_amdgcn_s_sleep(2);
      }
    }
  }
  __syncthreads();
  __threadfence();
}

// ---- persistent LSTM scan: 32 blocks x 128 threads, grid-barrier per step ----
// Block wg owns hidden cols [wg*16, wg*16+16). Wave g (0..3) computes gate g's
// 16x16 wmma tile: h_pad[16,512](bf16) x W_hh^T slice. W_hh slab held in VGPRs.
// The k-reduction uses 4 independent accumulator chains to shorten the per-step
// XDL dependency chain (the serial bottleneck across 1024 grid-synced steps).
__global__ __launch_bounds__(128) void lstm_scan_kernel(
    const float* __restrict__ xg,    // [M4, G4] precomputed x@W_ih^T + b_ih + b_hh
    const u16* __restrict__ Whh,     // [G4, H_] bf16
    u16* __restrict__ hseq,          // [M4, H_] bf16 output (layer activations)
    u16* __restrict__ hpad,          // [16, H_] bf16 shared h state (rows >=8 zero)
    unsigned* __restrict__ bar) {
  __shared__ float stage[4 * 8 * 16];   // [gate][batch<8][16 cols]

  const int tid = threadIdx.x;
  const int lane = tid & 31;
  const int gate = tid >> 5;            // wave id == gate id (i,f,g,o)
  const int hbase = blockIdx.x * 16;

  // Preload this wave's 16-column W_hh slab as 16 persistent B fragments (128 VGPRs)
  v16bf bfr[16];
#pragma unroll
  for (int kc = 0; kc < 16; ++kc) {
    const u16* p = Whh + (size_t)(gate * H_ + hbase + (lane & 15)) * H_ +
                   kc * 32 + ((lane >= 16) ? 16 : 0);
    FragBF f;
    f.u[0] = *(const uint4*)(p);
    f.u[1] = *(const uint4*)(p + 8);
    bfr[kc] = f.v;
  }

  // Zero this block's columns of hpad (all 16 padded rows)
  for (int i = tid; i < 256; i += 128) {
    int m = i >> 4, n = i & 15;
    hpad[m * H_ + hbase + n] = 0;
  }
  grid_barrier(bar);

  const int em = tid >> 4;              // 0..7  (batch)
  const int en = tid & 15;              // 0..15 (local hidden col)
  float creg = 0.f;                     // persistent cell state

  v8f zero = {0.f, 0.f, 0.f, 0.f, 0.f, 0.f, 0.f, 0.f};
  const int rofs = (lane >= 16) ? 8 : 0;

  for (int t = 0; t < T_; ++t) {
    // recurrent GEMM tile: [16pad x 512] x [512 x 16] for this wave's gate.
    // 4 independent wmma chains (4 k-chunks each), summed at the end.
    v8f pacc[4] = {zero, zero, zero, zero};
#pragma unroll
    for (int g = 0; g < 4; ++g) {
      v16bf a[4];
#pragma unroll
      for (int j = 0; j < 4; ++j) a[j] = load_a_frag(hpad, H_, 0, g * 4 + j);
#pragma unroll
      for (int j = 0; j < 4; ++j) {
        pacc[j] = __builtin_amdgcn_wmma_f32_16x16x32_bf16(
            false, a[j], false, bfr[g * 4 + j], (short)0, pacc[j], false, false);
      }
    }
    v8f acc = (pacc[0] + pacc[1]) + (pacc[2] + pacc[3]);

    // stage valid rows (batch 0..7) of this gate tile to LDS
    if (rofs == 0) {
      int n = lane & 15;
#pragma unroll
      for (int v = 0; v < 8; v++) stage[(gate * 8 + v) * 16 + n] = acc[v];
    }
    __syncthreads();

    // elementwise gate math: one thread per (batch, hidden col)
    {
      int hcol = hbase + en;
      size_t xrow = (size_t)(em * T_ + t) * G4;
      float gi = stage[(0 * 8 + em) * 16 + en] +
                 __builtin_nontemporal_load(&xg[xrow + 0 * H_ + hcol]);
      float gf = stage[(1 * 8 + em) * 16 + en] +
                 __builtin_nontemporal_load(&xg[xrow + 1 * H_ + hcol]);
      float gg = stage[(2 * 8 + em) * 16 + en] +
                 __builtin_nontemporal_load(&xg[xrow + 2 * H_ + hcol]);
      float go = stage[(3 * 8 + em) * 16 + en] +
                 __builtin_nontemporal_load(&xg[xrow + 3 * H_ + hcol]);
      gi = 1.f / (1.f + __expf(-gi));
      gf = 1.f / (1.f + __expf(-gf));
      gg = tanhf(gg);
      go = 1.f / (1.f + __expf(-go));
      creg = gf * creg + gi * gg;
      float h = go * tanhf(creg);
      u16 hb = f32_to_bf16(h);
      hpad[em * H_ + hcol] = hb;
      hseq[(size_t)(em * T_ + t) * H_ + hcol] = hb;
    }
    grid_barrier(bar);   // publish h before any block starts step t+1
  }
}

extern "C" void kernel_launch(void* const* d_in, const int* in_sizes, int n_in,
                              void* d_out, int out_size, void* d_ws, size_t ws_size,
                              hipStream_t stream) {
  const int*   ids   = (const int*)d_in[0];
  const float* emb   = (const float*)d_in[1];
  const float* Wih0  = (const float*)d_in[2];
  const float* Whh0  = (const float*)d_in[3];
  const float* bih0  = (const float*)d_in[4];
  const float* bhh0  = (const float*)d_in[5];
  const float* Wih1  = (const float*)d_in[6];
  const float* Whh1  = (const float*)d_in[7];
  const float* bih1  = (const float*)d_in[8];
  const float* bhh1  = (const float*)d_in[9];
  const float* Wfc   = (const float*)d_in[10];
  const float* bfc   = (const float*)d_in[11];
  float* out = (float*)d_out;

  // workspace carve-up (256B aligned)
  char* w = (char*)d_ws;
  auto alloc = [&](size_t bytes) {
    char* p = w;
    w += (bytes + 255) & ~(size_t)255;
    return p;
  };
  u16* Wfc_bf  = (u16*)alloc((size_t)V_ * H_ * 2);
  u16* Wih0_bf = (u16*)alloc((size_t)G4 * E_ * 2);
  u16* Whh0_bf = (u16*)alloc((size_t)G4 * H_ * 2);
  u16* Wih1_bf = (u16*)alloc((size_t)G4 * H_ * 2);
  u16* Whh1_bf = (u16*)alloc((size_t)G4 * H_ * 2);
  u16* x0_bf   = (u16*)alloc((size_t)M4 * E_ * 2);
  u16* h0seq   = (u16*)alloc((size_t)M4 * H_ * 2);
  u16* h1seq   = (u16*)alloc((size_t)M4 * H_ * 2);
  float* xg    = (float*)alloc((size_t)M4 * G4 * 4);
  u16* hpad    = (u16*)alloc((size_t)16 * H_ * 2);
  unsigned* bar = (unsigned*)alloc(256);

  hipMemsetAsync(bar, 0, 8, stream);

  auto cvt = [&](const float* src, u16* dst, int n) {
    int n4 = n / 4;
    cvt_bf16_kernel<<<(n4 + 255) / 256, 256, 0, stream>>>(src, (ushort4*)dst, n4);
  };
  cvt(Wfc,  Wfc_bf,  V_ * H_);
  cvt(Wih0, Wih0_bf, G4 * E_);
  cvt(Whh0, Whh0_bf, G4 * H_);
  cvt(Wih1, Wih1_bf, G4 * H_);
  cvt(Whh1, Whh1_bf, G4 * H_);

  // embedding gather -> bf16
  embed_kernel<<<(M4 * E_) / 256, 256, 0, stream>>>(ids, emb, x0_bf);

  // layer 0: xg = x0 @ W_ih0^T + b_ih0 + b_hh0
  gemm_bf16_kernel<false><<<dim3(G4 / 128, M4 / 128), 256, 0, stream>>>(
      x0_bf, Wih0_bf, bih0, bhh0, xg, M4, G4, E_);
  lstm_scan_kernel<<<32, 128, 0, stream>>>(xg, Whh0_bf, h0seq, hpad, bar);

  // layer 1: xg = h0 @ W_ih1^T + b_ih1 + b_hh1
  gemm_bf16_kernel<false><<<dim3(G4 / 128, M4 / 128), 256, 0, stream>>>(
      h0seq, Wih1_bf, bih1, bhh1, xg, M4, G4, H_);
  lstm_scan_kernel<<<32, 128, 0, stream>>>(xg, Whh1_bf, h1seq, hpad, bar);

  // FC: logits = h1 @ W_fc^T + b_fc  (store-bound; NT stores keep W_fc L2-resident)
  gemm_bf16_kernel<true><<<dim3(V_ / 128, M4 / 128), 256, 0, stream>>>(
      h1seq, Wfc_bf, bfc, nullptr, out, M4, V_, H_);
}
